// SmartCNN_70334384439333
// MI455X (gfx1250) — matrix operand order repaired and checked by reading, not compiled, not run on hardware
//
#include <hip/hip_runtime.h>

typedef _Float16 v16h __attribute__((ext_vector_type(16)));
typedef _Float16 h8   __attribute__((ext_vector_type(8)));
typedef float    v8f  __attribute__((ext_vector_type(8)));
typedef float    f4v  __attribute__((ext_vector_type(4)));

union AF { v16h v; h8 h[2]; _Float16 e[16]; };

#define Z1S 104   // padded stride (halves) for z1 [64 x 96]
#define Z2S 264   // padded stride (halves) for z2 [64 x 256]

static __device__ __forceinline__ v16h cvt16(const float* __restrict__ p) {
    AF f;
#pragma unroll
    for (int i = 0; i < 4; ++i) {
        f4v x = *(const f4v*)(p + (i >> 1) * 16 + (i & 1) * 4);
#pragma unroll
        for (int j = 0; j < 4; ++j) f.e[i * 4 + j] = (_Float16)x[j];
    }
    return f.v;
}

__global__ __launch_bounds__(256) void policy_fused(
    const int*   __restrict__ expo,
    const float* __restrict__ c0w, const float* __restrict__ c0b,
    const float* __restrict__ c1w,
    const float* __restrict__ lw,  const float* __restrict__ lb,
    const float* __restrict__ ow,  const float* __restrict__ ob,
    float*       __restrict__ out)
{
    __shared__ _Float16 sZ1[64 * Z1S];   // 13312 B  layer-1 A activations (f16)
    __shared__ _Float16 sZ2[64 * Z2S];   // 33792 B  layer-2 A activations (f16)
    __shared__ float    sLog[64 * 4];    // logits
    __shared__ int      sFlag[64];       // flip flags
    __shared__ float    sW0[96], sW1[96], sB0[4];

    const int tid = threadIdx.x;
    const int blk = blockIdx.x;

    // ---- stage conv weight tables into LDS ----
    if (tid < 96)       sW0[tid]       = c0w[tid];
    else if (tid < 192) sW1[tid - 96]  = c1w[tid - 96];
    else if (tid < 196) sB0[tid - 192] = c0b[tid - 192];
    __syncthreads();

    // ---- phase A: decode board, flips, conv-as-lookup -> z1 (4 threads/board) ----
    {
        const int board = tid >> 2;       // 0..63
        const int o     = tid & 3;        // conv output channel
        const int g     = blk * 64 + board;
        const int4* ep  = (const int4*)(expo + g * 16);
        int4 q0 = ep[0], q1 = ep[1], q2 = ep[2], q3 = ep[3];
        int e[16] = { q0.x, q0.y, q0.z, q0.w, q1.x, q1.y, q1.z, q1.w,
                      q2.x, q2.y, q2.z, q2.w, q3.x, q3.y, q3.z, q3.w };

        // argmax over corners (first-max wins; monotone in exponent)
        int best = e[0], ix = 0;
        if (e[3]  > best) { best = e[3];  ix = 1; }
        if (e[12] > best) { best = e[12]; ix = 2; }
        if (e[15] > best) { best = e[15]; ix = 3; }
        const bool fv = ix >= 2;
        const bool fh = (ix & 1) != 0;

        if (fv) {
#pragma unroll
            for (int c = 0; c < 4; ++c) {
                int t = e[c];     e[c]     = e[12 + c]; e[12 + c] = t;
                int u = e[4 + c]; e[4 + c] = e[8 + c];  e[8 + c]  = u;
            }
        }
        if (fh) {
#pragma unroll
            for (int r = 0; r < 4; ++r) {
                int b = r * 4;
                int t = e[b];     e[b]     = e[b + 3]; e[b + 3] = t;
                int u = e[b + 1]; e[b + 1] = e[b + 2]; e[b + 2] = u;
            }
        }

        _Float16* z = sZ1 + board * Z1S;
        const float w0m = sB0[o] + sW0[o * 24] + sW0[o * 24 + 1];
#pragma unroll
        for (int r = 0; r < 4; ++r)
#pragma unroll
            for (int c = 0; c < 3; ++c) {
                float v = w0m + sW0[o * 24 + (1 + e[r * 4 + c]) * 2]
                              + sW0[o * 24 + (1 + e[r * 4 + c + 1]) * 2 + 1];
                z[o * 12 + r * 3 + c] = (_Float16)fmaxf(v, 0.f);
            }
        const float w1m = sW1[o * 24] + sW1[o * 24 + 1];
#pragma unroll
        for (int r = 0; r < 3; ++r)
#pragma unroll
            for (int c = 0; c < 4; ++c) {
                float v = w1m + sW1[o * 24 + (1 + e[r * 4 + c]) * 2]
                              + sW1[o * 24 + (1 + e[(r + 1) * 4 + c]) * 2 + 1];
                z[48 + o * 12 + r * 4 + c] = (_Float16)fmaxf(v, 0.f);
            }
        if (o == 0) sFlag[board] = (fv ? 1 : 0) | (fh ? 2 : 0);
    }
    __syncthreads();

    const int wave   = tid >> 5;       // 0..7
    const int lane   = tid & 31;
    const int laneN  = lane & 15;      // N column / M row selector
    const int laneHi = lane >> 4;      // half-wave select

    // ---- layer 1: z2 = relu(z1 @ lw^T + lb), WMMA 16x16x32 f16, K=96 ----
    {
        v16h  bf[2][3];
        float bias[2];
#pragma unroll
        for (int t = 0; t < 2; ++t) {
            const int n = (wave * 2 + t) * 16 + laneN;   // output channel
            bias[t] = lb[n];
            const float* wr = lw + n * 96;
#pragma unroll
            for (int kk = 0; kk < 3; ++kk)               // B frag: K = kb + laneHi*8 + {0..7,16..23}
                bf[t][kk] = cvt16(wr + kk * 32 + laneHi * 8);
        }
#pragma unroll
        for (int rt = 0; rt < 4; ++rt) {
            const int r0 = rt * 16;
            const _Float16* zr = sZ1 + (r0 + laneN) * Z1S;
            AF af[3];
#pragma unroll
            for (int kk = 0; kk < 3; ++kk) {             // A frag: K = kb + laneHi*8 + {0..7,16..23}
                const int kb = kk * 32 + laneHi * 8;
                af[kk].h[0] = *(const h8*)(zr + kb);
                af[kk].h[1] = *(const h8*)(zr + kb + 16);
            }
#pragma unroll
            for (int t = 0; t < 2; ++t) {
                v8f acc;
#pragma unroll
                for (int j = 0; j < 8; ++j) acc[j] = bias[t];
#pragma unroll
                for (int kk = 0; kk < 3; ++kk)
                    acc = __builtin_amdgcn_wmma_f32_16x16x32_f16(
                        false, af[kk].v, false, bf[t][kk], (short)0, acc, false, false);
                const int n = (wave * 2 + t) * 16 + laneN;
#pragma unroll
                for (int j = 0; j < 8; ++j) {
                    float v = fmaxf(acc[j], 0.f);
                    sZ2[(r0 + j + laneHi * 8) * Z2S + n] = (_Float16)v;
                }
            }
        }
    }
    __syncthreads();

    // ---- layer 2: logits = z2 @ ow^T + ob (N padded 4->16), K=256 ----
    if (wave < 4) {
        const int r0 = wave * 16;
        const int n  = laneN;
        v8f acc;
        const float bias = (n < 4) ? ob[n] : 0.f;
#pragma unroll
        for (int j = 0; j < 8; ++j) acc[j] = bias;
        const _Float16* zr = sZ2 + (r0 + laneN) * Z2S;
#pragma unroll
        for (int kk = 0; kk < 8; ++kk) {
            AF af;
            const int kb = kk * 32 + laneHi * 8;
            af.h[0] = *(const h8*)(zr + kb);
            af.h[1] = *(const h8*)(zr + kb + 16);
            AF bfr;
            if (n < 4) {
                bfr.v = cvt16(ow + n * 256 + kk * 32 + laneHi * 8);
            } else {
#pragma unroll
                for (int i = 0; i < 16; ++i) bfr.e[i] = (_Float16)0.f;
            }
            acc = __builtin_amdgcn_wmma_f32_16x16x32_f16(
                false, af.v, false, bfr.v, (short)0, acc, false, false);
        }
        if (n < 4) {
#pragma unroll
            for (int j = 0; j < 8; ++j)
                sLog[(r0 + j + laneHi * 8) * 4 + n] = acc[j];
        }
    }
    __syncthreads();

    // ---- softmax + flip-unscramble + store (1 thread/board) ----
    if (tid < 64) {
        float l0 = sLog[tid * 4 + 0], l1 = sLog[tid * 4 + 1];
        float l2 = sLog[tid * 4 + 2], l3 = sLog[tid * 4 + 3];
        float m  = fmaxf(fmaxf(l0, l1), fmaxf(l2, l3));
        float e0 = __expf(l0 - m), e1 = __expf(l1 - m);
        float e2 = __expf(l2 - m), e3 = __expf(l3 - m);
        float inv = 1.f / (e0 + e1 + e2 + e3);
        float p0 = e0 * inv, p1 = e1 * inv, p2 = e2 * inv, p3 = e3 * inv;
        const int fl = sFlag[tid];
        const bool fv = (fl & 1) != 0, fh = (fl & 2) != 0;
        float4 r;
        r.x = fv ? p1 : p0;
        r.y = fv ? p0 : p1;
        r.z = fh ? p3 : p2;
        r.w = fh ? p2 : p3;
        ((float4*)out)[blk * 64 + tid] = r;
    }
}

extern "C" void kernel_launch(void* const* d_in, const int* in_sizes, int n_in,
                              void* d_out, int out_size, void* d_ws, size_t ws_size,
                              hipStream_t stream) {
    const int*   expo = (const int*)  d_in[0];
    const float* c0w  = (const float*)d_in[1];
    const float* c0b  = (const float*)d_in[2];
    const float* c1w  = (const float*)d_in[3];
    const float* lw   = (const float*)d_in[4];
    const float* lb   = (const float*)d_in[5];
    const float* ow   = (const float*)d_in[6];
    const float* ob   = (const float*)d_in[7];
    float* out = (float*)d_out;

    const int B      = in_sizes[0] / 16;   // 262144 boards
    const int blocks = B / 64;             // 64 boards per block
    policy_fused<<<blocks, 256, 0, stream>>>(expo, c0w, c0b, c1w, lw, lb, ow, ob, out);
}
